// MultiHeadAttention_88381837017790
// MI455X (gfx1250) — matrix operand order repaired and checked by reading, not compiled, not run on hardware
//
#include <hip/hip_runtime.h>

#define DEV static __device__ __forceinline__

#if defined(__has_builtin)
#if __has_builtin(__builtin_amdgcn_global_load_async_to_lds_b128) && \
    __has_builtin(__builtin_amdgcn_s_wait_asynccnt)
#define USE_ASYNC_LDS 1
#endif
#endif
#ifndef USE_ASYNC_LDS
#define USE_ASYNC_LDS 0
#endif

typedef __attribute__((ext_vector_type(16))) __bf16 v16bf;
typedef __attribute__((ext_vector_type(8)))  float  v8f;
typedef __attribute__((ext_vector_type(4)))  int    v4i;

namespace {
constexpr int Bc = 4, Sc = 2048, Dc = 1024, Hc = 16, DKc = 64;
constexpr int Mr = Bc * Sc;                              // 8192 GEMM rows
constexpr long long YELEMS = (long long)Bc * Sc * Dc;    // y region of d_out
constexpr long long ROWS   = (long long)Bc * Hc * Sc;    // attention rows
constexpr float SCALE = 0.125f;                          // 1/sqrt(DK)
constexpr float NEGBIG = -1.0e9f;
}

DEV unsigned short f2bf(float f) {
  unsigned u = __builtin_bit_cast(unsigned, f);
  u += 0x7FFFu + ((u >> 16) & 1u);                       // RNE
  return (unsigned short)(u >> 16);
}

DEV v8f wmma_bf16(v16bf a, v16bf b, v8f c) {
  return __builtin_amdgcn_wmma_f32_16x16x32_bf16(false, a, false, b, (short)0, c,
                                                 false, false);
}

// A fragment: 16x32 bf16 tile (row-major, leading dim ld).
DEV v16bf load_a_frag(const unsigned short* base, int ld, int lane) {
  int r  = lane & 15;
  int ko = (lane & 16) ? 8 : 0;
  const unsigned short* p = base + (size_t)r * ld + ko;
  union { uint4 u[2]; v16bf v; } c;
  c.u[0] = *(const uint4*)(p);
  c.u[1] = *(const uint4*)(p + 16);
  return c.v;
}

// B fragment: 32x16 bf16 tile from B^T (row n holds K contiguous, leading dim ld).
DEV v16bf load_b_frag(const unsigned short* baseT, int ld, int lane) {
  int n  = lane & 15;
  int ko = (lane & 16) ? 8 : 0;
  const unsigned short* p = baseT + (size_t)n * ld + ko;
  union { uint4 u[2]; v16bf v; } c;
  c.u[0] = *(const uint4*)(p);
  c.u[1] = *(const uint4*)(p + 16);
  return c.v;
}

// 16-byte global -> LDS copy (async path if available).
DEV void cp16_async(const unsigned short* src, unsigned short* dst) {
#if USE_ASYNC_LDS
  __builtin_amdgcn_global_load_async_to_lds_b128(
      (__attribute__((address_space(1))) v4i*)(src),
      (__attribute__((address_space(3))) v4i*)(dst), 0, 0);
#else
  (void)src; (void)dst;
#endif
}
DEV void async_drain() {
#if USE_ASYNC_LDS
  __builtin_amdgcn_s_wait_asynccnt(0);
#endif
}

DEV float wave_max(float v) {
  #pragma unroll
  for (int o = 16; o; o >>= 1) v = fmaxf(v, __shfl_xor(v, o, 32));
  return v;
}
DEV float wave_sum(float v) {
  #pragma unroll
  for (int o = 16; o; o >>= 1) v += __shfl_xor(v, o, 32);
  return v;
}

// ---------------- casts ----------------
__global__ void k_cast(const float* __restrict__ src, unsigned short* __restrict__ dst,
                       long long n4) {                   // n4 = n/4
  for (long long i = (long long)blockIdx.x * blockDim.x + threadIdx.x; i < n4;
       i += (long long)gridDim.x * blockDim.x) {
    float4 f = ((const float4*)src)[i];
    ushort4 o;
    o.x = f2bf(f.x); o.y = f2bf(f.y); o.z = f2bf(f.z); o.w = f2bf(f.w);
    ((ushort4*)dst)[i] = o;
  }
}

__global__ void k_cast_t(const float* __restrict__ W, unsigned short* __restrict__ Wt,
                         int K, int N) {
  long long total = (long long)K * N;
  for (long long t = (long long)blockIdx.x * blockDim.x + threadIdx.x; t < total;
       t += (long long)gridDim.x * blockDim.x) {
    int n = (int)(t / K), k = (int)(t % K);
    Wt[t] = f2bf(W[(size_t)k * N + n]);
  }
}

// ---------------- dense GEMM core: block tile 256(M) x 64(N), bf16 ----------------
// 8 waves/block; each wave owns 32x64. B tile (64x32) double-buffered in LDS.
struct GemmAcc { v8f a[2][4]; };

DEV void gemm_core(const unsigned short* __restrict__ A,
                   const unsigned short* __restrict__ Bt,
                   unsigned short* bsm /* [2][64*32] shared */,
                   int blockM, int tileN, int wave, int lane, int tid,
                   GemmAcc& acc) {
  #pragma unroll
  for (int i = 0; i < 2; ++i)
    #pragma unroll
    for (int j = 0; j < 4; ++j) acc.a[i][j] = (v8f){0,0,0,0,0,0,0,0};

  const int bn = tid >> 2, bc = (tid & 3) * 8;           // staging split: 16B/thread
  const unsigned short* bsrc0 = Bt + (size_t)(tileN + bn) * Dc + bc;
  unsigned short* bdst0 = bsm + bn * 32 + bc;
  const int rows = blockM + wave * 32;
  const unsigned short* a0p = A + (size_t)rows * Dc;
  const unsigned short* a1p = A + (size_t)(rows + 16) * Dc;

  // prologue: stage k-step 0 into buffer 0
#if USE_ASYNC_LDS
  cp16_async(bsrc0, bdst0);
  async_drain();
#else
  *(uint4*)bdst0 = *(const uint4*)bsrc0;
#endif
  __syncthreads();
  v16bf a0 = load_a_frag(a0p, Dc, lane);
  v16bf a1 = load_a_frag(a1p, Dc, lane);

  for (int ks = 0; ks < Dc / 32; ++ks) {
    unsigned short* bufc = bsm + (ks & 1) * (64 * 32);
    unsigned short* bufn = bsm + ((ks & 1) ^ 1) * (64 * 32);
    bool more = (ks + 1) < Dc / 32;
    v16bf a0n = a0, a1n = a1;
#if !USE_ASYNC_LDS
    uint4 breg = {};
#endif
    if (more) {
      int k1 = (ks + 1) * 32;
      a0n = load_a_frag(a0p + k1, Dc, lane);
      a1n = load_a_frag(a1p + k1, Dc, lane);
#if USE_ASYNC_LDS
      cp16_async(bsrc0 + k1, bufn + bn * 32 + bc);
#else
      breg = *(const uint4*)(bsrc0 + k1);
#endif
    }
    #pragma unroll
    for (int j = 0; j < 4; ++j) {
      v16bf b = load_b_frag(bufc + (j * 16) * 32, 32, lane);
      acc.a[0][j] = wmma_bf16(a0, b, acc.a[0][j]);
      acc.a[1][j] = wmma_bf16(a1, b, acc.a[1][j]);
    }
#if USE_ASYNC_LDS
    async_drain();
#else
    if (more) *(uint4*)(bufn + bn * 32 + bc) = breg;
#endif
    __syncthreads();
    a0 = a0n; a1 = a1n;
  }
}

__global__ void k_gemm_proj(const unsigned short* __restrict__ A,
                            const unsigned short* __restrict__ Bt,
                            const float* __restrict__ bias,
                            unsigned short* __restrict__ Out, int out_t) {
  __shared__ unsigned short bsm[2][64 * 32];
  int tid = threadIdx.x, wave = tid >> 5, lane = tid & 31;
  constexpr int MB = Mr / 256;                           // 32
  int blockM = (blockIdx.x % MB) * 256;
  int tileN  = (blockIdx.x / MB) * 64;
  GemmAcc acc;
  gemm_core(A, Bt, &bsm[0][0], blockM, tileN, wave, lane, tid, acc);

  int hi8 = (lane & 16) ? 8 : 0, nl = lane & 15;
  #pragma unroll
  for (int i = 0; i < 2; ++i) {
    int rbase = blockM + wave * 32 + i * 16 + hi8;
    #pragma unroll
    for (int j = 0; j < 4; ++j) {
      int col = tileN + j * 16 + nl;
      float bv = bias[col];
      #pragma unroll
      for (int r = 0; r < 8; ++r) {
        int m = rbase + r;
        float v = acc.a[i][j][r] + bv;
        if (!out_t) {
          Out[(size_t)m * Dc + col] = f2bf(v);
        } else {
          int bb = m / Sc, s = m % Sc;
          Out[((size_t)bb * Dc + col) * Sc + s] = f2bf(v);
        }
      }
    }
  }
}

__global__ void k_gemm_out(const unsigned short* __restrict__ A,
                           const unsigned short* __restrict__ Bt,
                           const float* __restrict__ bias,
                           const float* __restrict__ resid,
                           float* __restrict__ X) {
  __shared__ unsigned short bsm[2][64 * 32];
  int tid = threadIdx.x, wave = tid >> 5, lane = tid & 31;
  constexpr int MB = Mr / 256;
  int blockM = (blockIdx.x % MB) * 256;
  int tileN  = (blockIdx.x / MB) * 64;
  GemmAcc acc;
  gemm_core(A, Bt, &bsm[0][0], blockM, tileN, wave, lane, tid, acc);

  int hi8 = (lane & 16) ? 8 : 0, nl = lane & 15;
  #pragma unroll
  for (int i = 0; i < 2; ++i) {
    int rbase = blockM + wave * 32 + i * 16 + hi8;
    #pragma unroll
    for (int j = 0; j < 4; ++j) {
      int col = tileN + j * 16 + nl;
      float bv = bias[col];
      #pragma unroll
      for (int r = 0; r < 8; ++r) {
        int m = rbase + r;
        X[(size_t)m * Dc + col] = acc.a[i][j][r] + bv + resid[(size_t)m * Dc + col];
      }
    }
  }
}

// ---------------- scores: QK^T*scale + causal mask (K tile shared via LDS) --------
// Block = 8 waves, same (b,h,keyTile), 8 consecutive q-tiles.
__global__ void k_scores(const unsigned short* __restrict__ qp,
                         const unsigned short* __restrict__ kp,
                         float* __restrict__ attn) {
  __shared__ unsigned short ks_s[64 * 64];               // 64 keys x 64 dk
  int tid = threadIdx.x, wave = tid >> 5, lane = tid & 31;
  constexpr int KT = Sc / 64, QG = (Sc / 16) / 8;        // 32, 16
  int bx = blockIdx.x;
  int kt = bx % KT, qg = (bx / KT) % QG;
  int h = (bx / (KT * QG)) % Hc, b = bx / (KT * QG * Hc);
  int keyBase = kt * 64;
  int tileQ = (qg * 8 + wave) * 16;
  int tileQmax = (qg * 8 + 7) * 16;
  bool groupLive = keyBase <= tileQmax + 15;

  if (groupLive) {                                       // stage K tile (8KB)
    int key = tid >> 2, d0 = (tid & 3) * 16;
    const unsigned short* s = kp + (size_t)(b * Sc + keyBase + key) * Dc + h * DKc + d0;
    unsigned short* d = ks_s + key * 64 + d0;
#if USE_ASYNC_LDS
    cp16_async(s, d);
    cp16_async(s + 8, d + 8);
    async_drain();
#else
    *(uint4*)d = *(const uint4*)s;
    *(uint4*)(d + 8) = *(const uint4*)(s + 8);
#endif
  }
  __syncthreads();

  size_t rowBase = ((size_t)(b * Hc + h) * Sc + tileQ) * Sc;
  if (keyBase > tileQ + 15) {                            // this wave fully masked
    for (int idx = lane; idx < 16 * 64; idx += 32) {
      int rr = idx >> 6, kk = idx & 63;
      attn[rowBase + (size_t)rr * Sc + keyBase + kk] = NEGBIG;
    }
    return;
  }
  const unsigned short* qbase = qp + (size_t)(b * Sc + tileQ) * Dc + h * DKc;
  v16bf qa0 = load_a_frag(qbase, Dc, lane);
  v16bf qa1 = load_a_frag(qbase + 32, Dc, lane);
  v8f acc[4];
  #pragma unroll
  for (int j = 0; j < 4; ++j) acc[j] = (v8f){0,0,0,0,0,0,0,0};
  #pragma unroll
  for (int j = 0; j < 4; ++j) {
    acc[j] = wmma_bf16(qa0, load_b_frag(ks_s + (j * 16) * 64, 64, lane), acc[j]);
    acc[j] = wmma_bf16(qa1, load_b_frag(ks_s + (j * 16) * 64 + 32, 64, lane), acc[j]);
  }
  int hi8 = (lane & 16) ? 8 : 0, nl = lane & 15;
  #pragma unroll
  for (int j = 0; j < 4; ++j) {
    int key = keyBase + j * 16 + nl;
    #pragma unroll
    for (int r = 0; r < 8; ++r) {
      int q = tileQ + hi8 + r;
      float s = (key > q) ? NEGBIG : acc[j][r] * SCALE;
      attn[rowBase + (size_t)(hi8 + r) * Sc + key] = s;
    }
  }
}

// ---------------- per-row max / sum(exp), float4 ----------------
__global__ void k_rowstats(const float* __restrict__ attn, float* __restrict__ stats) {
  int tid = blockIdx.x * blockDim.x + threadIdx.x;
  int wid = tid >> 5, lane = tid & 31;
  if (wid >= ROWS) return;
  const float4* row = (const float4*)(attn + (size_t)wid * Sc);
  float m = -3.0e38f;
  for (int i = lane; i < Sc / 4; i += 32) {
    float4 f = row[i];
    m = fmaxf(m, fmaxf(fmaxf(f.x, f.y), fmaxf(f.z, f.w)));
  }
  m = wave_max(m);
  float s = 0.0f;
  for (int i = lane; i < Sc / 4; i += 32) {
    float4 f = row[i];
    s += __expf(f.x - m) + __expf(f.y - m) + __expf(f.z - m) + __expf(f.w - m);
  }
  s = wave_sum(s);
  if (lane == 0) { stats[wid] = m; stats[ROWS + wid] = s; }
}

// ---------------- normalize attn in place + O = P @ V (V tile shared) ----------
// Block = 8 waves, same (b,h), 8 consecutive q-tiles; V tile double-buffered.
__global__ void k_pv(const float* __restrict__ stats, float* __restrict__ attn,
                     const unsigned short* __restrict__ vt,
                     unsigned short* __restrict__ ctxh) {
  __shared__ unsigned short v_s[2][32 * 64];             // 2 x (32 keys x 64 dv)
  int tid = threadIdx.x, wave = tid >> 5, lane = tid & 31;
  constexpr int QG = (Sc / 16) / 8;                      // 16
  int bx = blockIdx.x;
  int qg = bx % QG, h = (bx / QG) % Hc, b = bx / (QG * Hc);
  int tileQ = (qg * 8 + wave) * 16;
  int kbZero = (tileQ + 16 + 31) & ~31;                  // first all-masked block
  int kbMaxG = (((qg * 8 + 7) * 16) + 16 + 31) & ~31;    // block-uniform bound

  int lrow = lane & 15, ko = (lane & 16) ? 8 : 0;
  size_t rowLin = (size_t)(b * Hc + h) * Sc + tileQ + lrow;
  float m = stats[rowLin];
  float invl = 1.0f / stats[ROWS + rowLin];
  float* arow = attn + rowLin * Sc;

  // staging split: 64 dv-rows x 32 keys, 16B per thread
  int vn = tid >> 2, vc = (tid & 3) * 8;
  const unsigned short* vsrc = vt + (size_t)(b * Dc + h * DKc + vn) * Sc + vc;
  v8f acc[4];
  #pragma unroll
  for (int j = 0; j < 4; ++j) acc[j] = (v8f){0,0,0,0,0,0,0,0};

  // prologue: stage kb=0
#if USE_ASYNC_LDS
  cp16_async(vsrc, &v_s[0][0] + vn * 32 + vc);
  async_drain();
#else
  *(uint4*)(&v_s[0][0] + vn * 32 + vc) = *(const uint4*)vsrc;
#endif
  __syncthreads();

  for (int kb = 0; kb < kbMaxG; kb += 32) {
    unsigned short* bufc = &v_s[(kb >> 5) & 1][0];
    unsigned short* bufn = &v_s[((kb >> 5) & 1) ^ 1][0];
    bool more = (kb + 32) < kbMaxG;
#if !USE_ASYNC_LDS
    uint4 vreg = {};
#endif
    if (more) {
#if USE_ASYNC_LDS
      cp16_async(vsrc + kb + 32, bufn + vn * 32 + vc);
#else
      vreg = *(const uint4*)(vsrc + kb + 32);
#endif
    }
    float* p = arow + kb + ko;
    if (kb < kbZero) {                                   // live block for this wave
      float rb[16];
      *(float4*)&rb[0]  = *(const float4*)(p);
      *(float4*)&rb[4]  = *(const float4*)(p + 4);
      *(float4*)&rb[8]  = *(const float4*)(p + 16);
      *(float4*)&rb[12] = *(const float4*)(p + 20);
      union { unsigned short s[16]; v16bf v; } pa;
      #pragma unroll
      for (int i = 0; i < 16; ++i) {
        float e = __expf(rb[i] - m) * invl;              // masked -> 0
        rb[i] = e;
        pa.s[i] = f2bf(e);
      }
      *(float4*)(p)      = *(const float4*)&rb[0];
      *(float4*)(p + 4)  = *(const float4*)&rb[4];
      *(float4*)(p + 16) = *(const float4*)&rb[8];
      *(float4*)(p + 20) = *(const float4*)&rb[12];
      #pragma unroll
      for (int j = 0; j < 4; ++j)
        acc[j] = wmma_bf16(pa.v, load_b_frag(bufc + (j * 16) * 32, 32, lane), acc[j]);
    } else {                                             // masked: required zeros
      float4 z = make_float4(0.f, 0.f, 0.f, 0.f);
      *(float4*)(p) = z; *(float4*)(p + 4) = z;
      *(float4*)(p + 16) = z; *(float4*)(p + 20) = z;
    }
#if USE_ASYNC_LDS
    async_drain();
#else
    if (more) *(uint4*)(bufn + vn * 32 + vc) = vreg;
#endif
    __syncthreads();
  }
  for (int kb = kbMaxG; kb < Sc; kb += 32) {             // tail: all masked
    float* p = arow + kb + ko;
    float4 z = make_float4(0.f, 0.f, 0.f, 0.f);
    *(float4*)(p) = z; *(float4*)(p + 4) = z;
    *(float4*)(p + 16) = z; *(float4*)(p + 20) = z;
  }

  int hi8 = (lane & 16) ? 8 : 0, nl = lane & 15;
  #pragma unroll
  for (int j = 0; j < 4; ++j) {
    int col = h * DKc + j * 16 + nl;
    #pragma unroll
    for (int r = 0; r < 8; ++r) {
      int s = tileQ + hi8 + r;
      ctxh[((size_t)b * Sc + s) * Dc + col] = f2bf(acc[j][r]);
    }
  }
}

// ---------------- LayerNorm: one wave per row, float4 ----------------
__global__ void k_layernorm(const float* __restrict__ X, const float* __restrict__ g,
                            const float* __restrict__ bta, float* __restrict__ Y) {
  int tid = blockIdx.x * blockDim.x + threadIdx.x;
  int wid = tid >> 5, lane = tid & 31;
  if (wid >= Mr) return;
  const float4* row = (const float4*)(X + (size_t)wid * Dc);
  float s1 = 0.f, s2 = 0.f;
  for (int i = lane; i < Dc / 4; i += 32) {
    float4 f = row[i];
    s1 += f.x + f.y + f.z + f.w;
    s2 += f.x * f.x + f.y * f.y + f.z * f.z + f.w * f.w;
  }
  s1 = wave_sum(s1); s2 = wave_sum(s2);
  float mu = s1 * (1.0f / Dc);
  float var = s2 * (1.0f / Dc) - mu * mu;
  float rstd = rsqrtf(var + 1e-5f);
  float4* out = (float4*)(Y + (size_t)wid * Dc);
  const float4* g4 = (const float4*)g;
  const float4* b4 = (const float4*)bta;
  for (int i = lane; i < Dc / 4; i += 32) {
    float4 f = row[i], gg = g4[i], bb = b4[i], o;
    o.x = (f.x - mu) * rstd * gg.x + bb.x;
    o.y = (f.y - mu) * rstd * gg.y + bb.y;
    o.z = (f.z - mu) * rstd * gg.z + bb.z;
    o.w = (f.w - mu) * rstd * gg.w + bb.w;
    out[i] = o;
  }
}

extern "C" void kernel_launch(void* const* d_in, const int* in_sizes, int n_in,
                              void* d_out, int out_size, void* d_ws, size_t ws_size,
                              hipStream_t stream) {
  (void)in_sizes; (void)n_in; (void)out_size; (void)ws_size;
  const float* Qf  = (const float*)d_in[0];
  const float* Kf  = (const float*)d_in[1];
  const float* Vf  = (const float*)d_in[2];
  const float* W_Q = (const float*)d_in[4];  const float* b_Q = (const float*)d_in[5];
  const float* W_K = (const float*)d_in[6];  const float* b_K = (const float*)d_in[7];
  const float* W_V = (const float*)d_in[8];  const float* b_V = (const float*)d_in[9];
  const float* W_O = (const float*)d_in[10]; const float* b_O = (const float*)d_in[11];
  const float* gam = (const float*)d_in[12]; const float* bet = (const float*)d_in[13];

  float* y    = (float*)d_out;
  float* attn = y + YELEMS;

  size_t off = 0;
  auto take = [&](size_t bytes) -> char* {
    char* p = (char*)d_ws + off;
    off += (bytes + 255) & ~(size_t)255;
    return p;
  };
  unsigned short* Qc  = (unsigned short*)take(YELEMS * 2);
  unsigned short* Kc  = (unsigned short*)take(YELEMS * 2);
  unsigned short* Vc  = (unsigned short*)take(YELEMS * 2);
  unsigned short* Wqt = (unsigned short*)take((size_t)Dc * Dc * 2);
  unsigned short* Wkt = (unsigned short*)take((size_t)Dc * Dc * 2);
  unsigned short* Wvt = (unsigned short*)take((size_t)Dc * Dc * 2);
  unsigned short* Wot = (unsigned short*)take((size_t)Dc * Dc * 2);
  unsigned short* qp  = (unsigned short*)take(YELEMS * 2);
  unsigned short* kp  = (unsigned short*)take(YELEMS * 2);
  unsigned short* vt  = (unsigned short*)take(YELEMS * 2);
  float*          st  = (float*)take((size_t)ROWS * 2 * 4);
  unsigned short* ctxh = Qc;                     // Qc dead after q projection
  float*          X    = (float*)Kc;            // Kc+Vc dead after k/v proj

  const int T = 256;
  k_cast<<<2048, T, 0, stream>>>(Qf, Qc, YELEMS / 4);
  k_cast<<<2048, T, 0, stream>>>(Kf, Kc, YELEMS / 4);
  k_cast<<<2048, T, 0, stream>>>(Vf, Vc, YELEMS / 4);
  k_cast_t<<<1024, T, 0, stream>>>(W_Q, Wqt, Dc, Dc);
  k_cast_t<<<1024, T, 0, stream>>>(W_K, Wkt, Dc, Dc);
  k_cast_t<<<1024, T, 0, stream>>>(W_V, Wvt, Dc, Dc);
  k_cast_t<<<1024, T, 0, stream>>>(W_O, Wot, Dc, Dc);

  const int gemmBlocks = (Mr / 256) * (Dc / 64);         // 512
  k_gemm_proj<<<gemmBlocks, T, 0, stream>>>(Qc, Wqt, b_Q, qp, 0);
  k_gemm_proj<<<gemmBlocks, T, 0, stream>>>(Kc, Wkt, b_K, kp, 0);
  k_gemm_proj<<<gemmBlocks, T, 0, stream>>>(Vc, Wvt, b_V, vt, 1);

  k_scores<<<Bc * Hc * ((Sc / 16) / 8) * (Sc / 64), T, 0, stream>>>(qp, kp, attn);
  k_rowstats<<<(int)(ROWS / 8), T, 0, stream>>>(attn, st);
  k_pv<<<Bc * Hc * ((Sc / 16) / 8), T, 0, stream>>>(st, attn, vt, ctxh);

  k_gemm_out<<<gemmBlocks, T, 0, stream>>>(ctxh, Wot, b_O, Qf, X);
  k_layernorm<<<Mr / 8, T, 0, stream>>>(X, gam, bet, y);
}
